// MACE_44203803410588
// MI455X (gfx1250) — matrix operand order repaired and checked by reading, not compiled, not run on hardware
//
#include <hip/hip_runtime.h>

typedef __attribute__((ext_vector_type(16))) _Float16 v16h;
typedef __attribute__((ext_vector_type(8)))  float    v8f;

// f16 weight buffer offsets (in halves), transposed layouts Wt[n][k]
#define WNL0T 0            // [64][64]
#define WNL1T 4096         // [64][64]
#define WM1T  8192         // [64][32] (K padded 8->32 with zeros)
#define WM2T  10240        // [64][64]
#define WM3T  14336        // [320][64]
#define W0T   34816        // [64][128]
#define W1T   43008        // [64][192]

// ---- WMMA fragment index helpers (per CDNA5 ISA 7.12.2, wave32) ----
// A (16x32 f16): lane holds row M=lane&15; lanes<16 cover K {0..7,16..23},
// lanes>=16 cover K {8..15,24..31}; VGPR v packs K pairs.
__device__ __forceinline__ int a_kidx(int lane, int idx) {
  return ((lane & 16) >> 1) + ((idx & 8) << 1) + (idx & 7);
}
// B (32x16 f16): lane holds column N=lane&15; lanes<16 K {0..7,16..23}, lanes>=16 K {8..15,24..31}.
__device__ __forceinline__ int b_kidx(int lane, int idx) {
  return ((lane & 16) >> 1) + ((idx & 8) << 1) + (idx & 7);
}
__device__ __forceinline__ v8f wmma32f16(v16h a, v16h b, v8f c) {
  return __builtin_amdgcn_wmma_f32_16x16x32_f16(false, a, false, b, (short)0, c,
                                                false, false);
}
__device__ __forceinline__ float silu_f(float x) {
  return x / (1.0f + __expf(-x));
}
// Wave-local LDS sync: raw CDNA5 counter wait (no clang builtin for dscnt on
// this toolchain) + scheduling barrier; "memory" clobber stops reordering.
__device__ __forceinline__ void wave_lds_sync() {
  asm volatile("s_wait_dscnt 0" ::: "memory");
  __builtin_amdgcn_wave_barrier();
}
__device__ __forceinline__ v16h load_a_lds(const _Float16* base, int lane,
                                           int ldk, int koff) {
  int row = lane & 15;
  v16h a;
#pragma unroll
  for (int idx = 0; idx < 16; ++idx)
    a[idx] = base[row * ldk + koff + a_kidx(lane, idx)];
  return a;
}
__device__ __forceinline__ v16h load_b_lds(const _Float16* base, int lane,
                                           int ldk, int nrow, int koff) {
  v16h b;
#pragma unroll
  for (int idx = 0; idx < 16; ++idx)
    b[idx] = base[nrow * ldk + koff + b_kidx(lane, idx)];
  return b;
}
__device__ __forceinline__ v16h load_a_gf32(const float* base, int lane,
                                            int ldk, int koff, float scale) {
  int row = lane & 15;
  const float* p = base + (size_t)row * ldk + koff;
  v16h a;
#pragma unroll
  for (int idx = 0; idx < 16; ++idx)
    a[idx] = (_Float16)(p[a_kidx(lane, idx)] * scale);
  return a;
}

// ---------------- kernel 0: zero the scatter accumulators ----------------
__global__ void mace_zero(float4* p, long long n4) {
  long long i = (long long)blockIdx.x * blockDim.x + threadIdx.x;
  long long stride = (long long)gridDim.x * blockDim.x;
  float4 z; z.x = z.y = z.z = z.w = 0.0f;
  for (; i < n4; i += stride) p[i] = z;
}

// ---------------- kernel 1: weights f32 -> f16, transposed ----------------
__global__ void mace_prep_weights(const float* Wnl0, const float* Wnl1,
                                  const float* Wm1, const float* Wm2,
                                  const float* Wm3, const float* W0,
                                  const float* W1, _Float16* wbuf) {
  int tid = blockIdx.x * blockDim.x + threadIdx.x;
  int stride = gridDim.x * blockDim.x;
  for (int i = tid; i < 64 * 64; i += stride) {
    int nn = i >> 6, k = i & 63;
    wbuf[WNL0T + i] = (_Float16)Wnl0[k * 64 + nn];
    wbuf[WNL1T + i] = (_Float16)Wnl1[k * 64 + nn];
    wbuf[WM2T + i] = (_Float16)Wm2[k * 64 + nn];
  }
  for (int i = tid; i < 64 * 32; i += stride) {
    int nn = i >> 5, k = i & 31;
    wbuf[WM1T + i] = (k < 8) ? (_Float16)Wm1[k * 64 + nn] : (_Float16)0.0f;
  }
  for (int i = tid; i < 320 * 64; i += stride) {
    int nn = i >> 6, k = i & 63;
    wbuf[WM3T + i] = (_Float16)Wm3[k * 320 + nn];
  }
  for (int i = tid; i < 64 * 128; i += stride) {
    int nn = i >> 7, k = i & 127;
    wbuf[W0T + i] = (_Float16)W0[k * 64 + nn];
  }
  for (int i = tid; i < 64 * 192; i += stride) {
    int nn = i / 192, k = i % 192;
    wbuf[W1T + i] = (_Float16)W1[k * 64 + nn];
  }
}

// --------- kernel 2: node linears h0=x0@Wnl0, h1=x1@Wnl1 (+ sc copy) -------
// H layout: H[(node*4 + comp)*64 + f], comp0=h0, comp1..3=h1[i]
__global__ void mace_node_linear(const float* __restrict__ node_feats,
                                 const _Float16* __restrict__ wbuf,
                                 float* __restrict__ H,
                                 float* __restrict__ sc_out, int ntiles) {
  __shared__ _Float16 sW[2][64 * 64];
  int tid = threadIdx.x;
  for (int i = tid; i < 64 * 64; i += blockDim.x) {
    sW[0][i] = wbuf[WNL0T + i];
    sW[1][i] = wbuf[WNL1T + i];
  }
  __syncthreads();
  int nwaves = blockDim.x >> 5;
  int wave = tid >> 5, lane = tid & 31;
  int n = lane & 15;
  int mhi = (lane & 16) >> 1;
  for (int tile = blockIdx.x * nwaves + wave; tile < ntiles;
       tile += gridDim.x * nwaves) {
    int n0 = tile * 16;
    const float* rows = node_feats + (size_t)n0 * 256;
#pragma unroll
    for (int comp = 0; comp < 4; ++comp) {
      const _Float16* Wt = sW[comp ? 1 : 0];
      v16h a0 = load_a_gf32(rows + comp * 64, lane, 256, 0, 1.0f);
      v16h a1 = load_a_gf32(rows + comp * 64, lane, 256, 32, 1.0f);
#pragma unroll
      for (int j = 0; j < 4; ++j) {
        v8f c;
#pragma unroll
        for (int v = 0; v < 8; ++v) c[v] = 0.0f;
        c = wmma32f16(a0, load_b_lds(Wt, lane, 64, j * 16 + n, 0), c);
        c = wmma32f16(a1, load_b_lds(Wt, lane, 64, j * 16 + n, 32), c);
#pragma unroll
        for (int v = 0; v < 8; ++v)
          H[((size_t)(n0 + v + mhi) * 4 + comp) * 64 + j * 16 + n] = c[v];
      }
    }
    const float4* srcp = (const float4*)rows;
    float4* dstp = (float4*)(sc_out + (size_t)n0 * 256);
    for (int i = lane; i < 16 * 64; i += 32) dstp[i] = srcp[i];
  }
}

// ---- kernel 3: fused edge MLP (WMMA) + tensor product + atomic scatter ----
__global__ void __launch_bounds__(128)
mace_edge(const float* __restrict__ edge_attrs,
          const float* __restrict__ edge_feats,
          const int* __restrict__ edge_index, const float* __restrict__ bm1,
          const float* __restrict__ bm2, const float* __restrict__ bm3,
          const _Float16* __restrict__ wbuf, const float* __restrict__ H,
          float* __restrict__ M0, float* __restrict__ M1, int ntiles) {
  __shared__ _Float16 sWm1[64 * 32];    // 4 KB
  __shared__ _Float16 sWm2[64 * 64];    // 8 KB
  __shared__ _Float16 sWm3[320 * 64];   // 40 KB
  __shared__ _Float16 stg[4][16 * 64];  // 8 KB, per-wave staging
  int tid = threadIdx.x;
  for (int i = tid; i < 64 * 32; i += blockDim.x) sWm1[i] = wbuf[WM1T + i];
  for (int i = tid; i < 64 * 64; i += blockDim.x) sWm2[i] = wbuf[WM2T + i];
  for (int i = tid; i < 320 * 64; i += blockDim.x) sWm3[i] = wbuf[WM3T + i];
  __syncthreads();
  int nwaves = blockDim.x >> 5;
  int wave = tid >> 5, lane = tid & 31;
  int n = lane & 15;
  int mhi = (lane & 16) >> 1;
  _Float16* S = stg[wave];
  const float inv_sq3 = 0.5773502691896258f;
  const float inv_sq2 = 0.7071067811865476f;
  for (int tile = blockIdx.x * nwaves + wave; tile < ntiles;
       tile += gridDim.x * nwaves) {
    int e0 = tile * 16;
    // stage layer-1 A tile (16 edges x K=32, K padded 8->32)
    for (int i = lane; i < 16 * 32; i += 32) S[i] = (_Float16)0.0f;
    for (int i = lane; i < 16 * 8; i += 32) {
      int r = i >> 3, k = i & 7;
      S[r * 32 + k] = (_Float16)edge_feats[(size_t)(e0 + r) * 8 + k];
    }
    wave_lds_sync();
    v16h a1 = load_a_lds(S, lane, 32, 0);
    float acc[4][8];
#pragma unroll
    for (int j = 0; j < 4; ++j) {
      float bv = bm1[j * 16 + n];
      v8f c;
#pragma unroll
      for (int v = 0; v < 8; ++v) c[v] = bv;
      c = wmma32f16(a1, load_b_lds(sWm1, lane, 32, j * 16 + n, 0), c);
#pragma unroll
      for (int v = 0; v < 8; ++v) acc[j][v] = silu_f(c[v]);
    }
    wave_lds_sync();
#pragma unroll
    for (int j = 0; j < 4; ++j)
#pragma unroll
      for (int v = 0; v < 8; ++v)
        S[(v + mhi) * 64 + j * 16 + n] = (_Float16)acc[j][v];
    wave_lds_sync();
    v16h a2lo = load_a_lds(S, lane, 64, 0);
    v16h a2hi = load_a_lds(S, lane, 64, 32);
#pragma unroll
    for (int j = 0; j < 4; ++j) {
      float bv = bm2[j * 16 + n];
      v8f c;
#pragma unroll
      for (int v = 0; v < 8; ++v) c[v] = bv;
      c = wmma32f16(a2lo, load_b_lds(sWm2, lane, 64, j * 16 + n, 0), c);
      c = wmma32f16(a2hi, load_b_lds(sWm2, lane, 64, j * 16 + n, 32), c);
#pragma unroll
      for (int v = 0; v < 8; ++v) acc[j][v] = silu_f(c[v]);
    }
    wave_lds_sync();
#pragma unroll
    for (int j = 0; j < 4; ++j)
#pragma unroll
      for (int v = 0; v < 8; ++v)
        S[(v + mhi) * 64 + j * 16 + n] = (_Float16)acc[j][v];
    wave_lds_sync();
    v16h a3lo = load_a_lds(S, lane, 64, 0);
    v16h a3hi = load_a_lds(S, lane, 64, 32);
    // per-lane edge metadata for its 8 edge rows
    int esrc[8], edst[8];
    float ey0[8], eyv[8][3];
#pragma unroll
    for (int v = 0; v < 8; ++v) {
      int e = e0 + v + mhi;
      esrc[v] = edge_index[2 * e];
      edst[v] = edge_index[2 * e + 1];
      ey0[v] = edge_attrs[4 * e];
      eyv[v][0] = edge_attrs[4 * e + 1];
      eyv[v][1] = edge_attrs[4 * e + 2];
      eyv[v][2] = edge_attrs[4 * e + 3];
    }
    // layer 3 in 4 feature blocks of 16; 5 weight groups each
    for (int b = 0; b < 4; ++b) {
      v8f w[5];
#pragma unroll
      for (int g = 0; g < 5; ++g) {
        int nn = (g * 4 + b) * 16 + n;
        float bv = bm3[nn];
        v8f c;
#pragma unroll
        for (int v = 0; v < 8; ++v) c[v] = bv;
        c = wmma32f16(a3lo, load_b_lds(sWm3, lane, 64, nn, 0), c);
        c = wmma32f16(a3hi, load_b_lds(sWm3, lane, 64, nn, 32), c);
        w[g] = c;
      }
      int f = b * 16 + n;
#pragma unroll
      for (int v = 0; v < 8; ++v) {
        const float* hs = H + (size_t)esrc[v] * 256;
        float x0 = hs[f];
        float x1[3] = {hs[64 + f], hs[128 + f], hs[192 + f]};
        float y0 = ey0[v];
        float ya = eyv[v][0], yb = eyv[v][1], yc = eyv[v][2];
        float w00 = w[0][v], w01 = w[1][v], w10 = w[2][v];
        float w110 = w[3][v], w111 = w[4][v];
        float ydot = ya * x1[0] + yb * x1[1] + yc * x1[2];
        float cr[3] = {yb * x1[2] - yc * x1[1], yc * x1[0] - ya * x1[2],
                       ya * x1[1] - yb * x1[0]};
        float yv[3] = {ya, yb, yc};
        float* M0d = M0 + (size_t)edst[v] * 128;
        atomicAdd(M0d + f, w00 * x0 * y0);
        atomicAdd(M0d + 64 + f, w110 * ydot * inv_sq3);
        float* M1d = M1 + (size_t)edst[v] * 576;  // [3][192]
#pragma unroll
        for (int i2 = 0; i2 < 3; ++i2) {
          atomicAdd(M1d + i2 * 192 + f, w01 * x0 * yv[i2]);
          atomicAdd(M1d + i2 * 192 + 64 + f, w10 * x1[i2] * y0);
          atomicAdd(M1d + i2 * 192 + 128 + f, w111 * cr[i2] * inv_sq2);
        }
      }
    }
  }
}

// ------------- kernel 4: output linears o0=(M0/16)@W0, o1=(M1/16)@W1 -------
__global__ void mace_out(const float* __restrict__ M0,
                         const float* __restrict__ M1,
                         const _Float16* __restrict__ wbuf,
                         float* __restrict__ out, int ntiles) {
  __shared__ _Float16 sW0[64 * 128];  // 16 KB
  __shared__ _Float16 sW1[64 * 192];  // 24 KB
  int tid = threadIdx.x;
  for (int i = tid; i < 64 * 128; i += blockDim.x) sW0[i] = wbuf[W0T + i];
  for (int i = tid; i < 64 * 192; i += blockDim.x) sW1[i] = wbuf[W1T + i];
  __syncthreads();
  int nwaves = blockDim.x >> 5;
  int wave = tid >> 5, lane = tid & 31;
  int n = lane & 15;
  int mhi = (lane & 16) >> 1;
  const float scale = 1.0f / 16.0f;  // / AVG_NEIGH
  for (int tile = blockIdx.x * nwaves + wave; tile < ntiles;
       tile += gridDim.x * nwaves) {
    int n0 = tile * 16;
    {
      v16h af[4];
#pragma unroll
      for (int kc = 0; kc < 4; ++kc)
        af[kc] = load_a_gf32(M0 + (size_t)n0 * 128, lane, 128, kc * 32, scale);
#pragma unroll
      for (int j = 0; j < 4; ++j) {
        v8f c;
#pragma unroll
        for (int v = 0; v < 8; ++v) c[v] = 0.0f;
#pragma unroll
        for (int kc = 0; kc < 4; ++kc)
          c = wmma32f16(af[kc], load_b_lds(sW0, lane, 128, j * 16 + n, kc * 32),
                        c);
#pragma unroll
        for (int v = 0; v < 8; ++v)
          out[(size_t)(n0 + v + mhi) * 256 + j * 16 + n] = c[v];
      }
    }
#pragma unroll
    for (int i2 = 0; i2 < 3; ++i2) {
      v16h ag[6];
#pragma unroll
      for (int kc = 0; kc < 6; ++kc)
        ag[kc] = load_a_gf32(M1 + ((size_t)n0 * 3 + i2) * 192, lane, 576,
                             kc * 32, scale);
#pragma unroll
      for (int j = 0; j < 4; ++j) {
        v8f c;
#pragma unroll
        for (int v = 0; v < 8; ++v) c[v] = 0.0f;
#pragma unroll
        for (int kc = 0; kc < 6; ++kc)
          c = wmma32f16(ag[kc], load_b_lds(sW1, lane, 192, j * 16 + n, kc * 32),
                        c);
#pragma unroll
        for (int v = 0; v < 8; ++v)
          out[(size_t)(n0 + v + mhi) * 256 + 64 + i2 * 64 + j * 16 + n] = c[v];
      }
    }
  }
}

extern "C" void kernel_launch(void* const* d_in, const int* in_sizes, int n_in,
                              void* d_out, int out_size, void* d_ws,
                              size_t ws_size, hipStream_t stream) {
  const float* node_feats = (const float*)d_in[0];
  const float* edge_attrs = (const float*)d_in[1];
  const float* edge_feats = (const float*)d_in[2];
  const int* edge_index = (const int*)d_in[3];
  const float* W_nl0 = (const float*)d_in[4];
  const float* W_nl1 = (const float*)d_in[5];
  const float* Wm1 = (const float*)d_in[6];
  const float* bm1 = (const float*)d_in[7];
  const float* Wm2 = (const float*)d_in[8];
  const float* bm2 = (const float*)d_in[9];
  const float* Wm3 = (const float*)d_in[10];
  const float* bm3 = (const float*)d_in[11];
  const float* W0 = (const float*)d_in[12];
  const float* W1 = (const float*)d_in[13];
  int Nn = in_sizes[0] / 256;  // nodes
  int Ee = in_sizes[3] / 2;    // edges
  // workspace layout: M0 [N][128] | M1 [N][3][192] | H [N][4][64] | wbuf f16
  float* M0 = (float*)d_ws;
  float* M1 = M0 + (size_t)Nn * 128;
  float* H = M0 + (size_t)Nn * 704;
  _Float16* wbuf = (_Float16*)(H + (size_t)Nn * 256);
  float* out = (float*)d_out;
  float* sc_out = out + (size_t)Nn * 256;

  long long zn4 = ((long long)Nn * 704) / 4;
  mace_zero<<<2048, 256, 0, stream>>>((float4*)M0, zn4);
  mace_prep_weights<<<64, 256, 0, stream>>>(W_nl0, W_nl1, Wm1, Wm2, Wm3, W0, W1,
                                            wbuf);
  int ntile_n = Nn / 16;
  int ntile_e = Ee / 16;
  int nlblocks = (ntile_n + 3) / 4;  // 4 waves/block, 1 tile each per step
  mace_node_linear<<<nlblocks, 128, 0, stream>>>(node_feats, wbuf, H, sc_out,
                                                 ntile_n);
  int eblocks = (ntile_e + 15) / 16;  // 4 waves/block, ~4 tiles per wave
  mace_edge<<<eblocks, 128, 0, stream>>>(edge_attrs, edge_feats, edge_index,
                                         bm1, bm2, bm3, wbuf, H, M0, M1,
                                         ntile_e);
  mace_out<<<nlblocks, 128, 0, stream>>>(M0, M1, wbuf, out, ntile_n);
}
